// TransformerLayer_33148557590904
// MI455X (gfx1250) — matrix operand order repaired and checked by reading, not compiled, not run on hardware
//
#include <hip/hip_runtime.h>
#include <hip/hip_bf16.h>
#include <math.h>
#include <stdint.h>

typedef __bf16 bf16_t;
typedef __attribute__((ext_vector_type(16))) __bf16 v16bf;
typedef __attribute__((ext_vector_type(8)))  __bf16 v8bf;
typedef __attribute__((ext_vector_type(8)))  float  v8f;
typedef __attribute__((ext_vector_type(4)))  unsigned int u32x4;
typedef __attribute__((ext_vector_type(8)))  int i32x8;
typedef __attribute__((ext_vector_type(4)))  int i32x4;

#define D_MODEL 1024
#define SEQ_T   2048
#define BATCH   4
#define M8      (BATCH * SEQ_T)   // 8192 merged rows

// ---------------- elementwise f32 -> bf16 weight conversion ----------------
__global__ void cvt_f32_bf16(const float* __restrict__ src,
                             bf16_t* __restrict__ dst, int n) {
  int i = blockIdx.x * 256 + threadIdx.x;
  if (i < n) dst[i] = (bf16_t)src[i];
}

// ---------------- LayerNorm (row = 1024 floats) -> bf16 ----------------
__global__ __launch_bounds__(256) void layernorm_bf16(
    const float* __restrict__ X, const float* __restrict__ g,
    const float* __restrict__ b, bf16_t* __restrict__ out) {
  __shared__ float red[256];
  const long row = blockIdx.x;
  const float* x = X + row * (long)D_MODEL;
  const int t = threadIdx.x;
  float4 v = ((const float4*)x)[t];
  red[t] = v.x + v.y + v.z + v.w;
  __syncthreads();
  for (int k = 128; k > 0; k >>= 1) {
    if (t < k) red[t] += red[t + k];
    __syncthreads();
  }
  const float mu = red[0] * (1.0f / D_MODEL);
  __syncthreads();
  const float dx = v.x - mu, dy = v.y - mu, dz = v.z - mu, dw = v.w - mu;
  red[t] = dx * dx + dy * dy + dz * dz + dw * dw;
  __syncthreads();
  for (int k = 128; k > 0; k >>= 1) {
    if (t < k) red[t] += red[t + k];
    __syncthreads();
  }
  const float rs = rsqrtf(red[0] * (1.0f / D_MODEL) + 1e-5f);
  const int c = 4 * t;
  bf16_t* o = out + row * (long)D_MODEL + c;
  o[0] = (bf16_t)(dx * rs * g[c + 0] + b[c + 0]);
  o[1] = (bf16_t)(dy * rs * g[c + 1] + b[c + 1]);
  o[2] = (bf16_t)(dz * rs * g[c + 2] + b[c + 2]);
  o[3] = (bf16_t)(dw * rs * g[c + 3] + b[c + 3]);
}

// ---------------- row softmax over T=2048 bf16 scores, in place ----------------
__global__ __launch_bounds__(256) void softmax_rows_bf16(bf16_t* __restrict__ S) {
  __shared__ float red[256];
  const long row = blockIdx.x;  // 0 .. B*T-1
  bf16_t* p = S + row * (long)SEQ_T;
  const int t = threadIdx.x;
  float v[8];
  float mx = -INFINITY;
#pragma unroll
  for (int i = 0; i < 8; ++i) {
    v[i] = (float)p[t + i * 256];
    mx = fmaxf(mx, v[i]);
  }
  red[t] = mx; __syncthreads();
  for (int k = 128; k > 0; k >>= 1) {
    if (t < k) red[t] = fmaxf(red[t], red[t + k]);
    __syncthreads();
  }
  mx = red[0]; __syncthreads();
  float s = 0.f;
#pragma unroll
  for (int i = 0; i < 8; ++i) { v[i] = __expf(v[i] - mx); s += v[i]; }
  red[t] = s; __syncthreads();
  for (int k = 128; k > 0; k >>= 1) {
    if (t < k) red[t] += red[t + k];
    __syncthreads();
  }
  const float inv = 1.0f / red[0];
#pragma unroll
  for (int i = 0; i < 8; ++i) p[t + i * 256] = (bf16_t)(v[i] * inv);
}

// ---------------- WMMA bf16 GEMM: C[M,N] = A[M,K] @ W[N,K]^T (+epilogues) ------
// Block tile 64(M) x 128(N), K tile 64, 128 threads = 4 waves, wave tile 32x64.
// A tile: per-lane GLOBAL_LOAD_ASYNC_TO_LDS_B128 (ASYNCcnt).
// B tile: one TDM descriptor per stage, issued by wave 0 (TENSORcnt).
#define BM 64
#define BN 128
#define KTILE 64
#define LPAD  8
#define LROW  (KTILE + LPAD)

#if defined(__has_builtin)
#if __has_builtin(__builtin_amdgcn_tensor_load_to_lds)
#define HAVE_TDM 1
#endif
#endif
#ifndef HAVE_TDM
#define HAVE_TDM 0
#endif

__device__ inline v16bf cat8(v8bf lo, v8bf hi) {
  return __builtin_shufflevector(lo, hi, 0, 1, 2, 3, 4, 5, 6, 7,
                                 8, 9, 10, 11, 12, 13, 14, 15);
}

__device__ inline void async_b128_to_lds(const bf16_t* gsrc, const bf16_t* lds_dst) {
  const uint64_t ga = (uint64_t)(uintptr_t)gsrc;
  const uint32_t la = (uint32_t)(uintptr_t)lds_dst;  // low 32b of flat LDS ptr = LDS offset
  asm volatile("global_load_async_to_lds_b128 %0, %1, off"
               :: "v"(la), "v"(ga) : "memory");
}

__device__ inline void wait_async0() {
  asm volatile("s_wait_asynccnt 0" ::: "memory");
}

// TDM: load a [rows x KTILE] bf16 tile (row stride kd elements) into LDS with
// 16B padding after every 128B row (reproduces the LROW = KTILE+8 LDS pitch).
__device__ inline void tdm_load_tile(const bf16_t* gsrc, const bf16_t* lds_dst,
                                     int rows, int kd) {
#if HAVE_TDM
  const uint64_t ga = (uint64_t)(uintptr_t)gsrc;
  const uint32_t la = (uint32_t)(uintptr_t)lds_dst;
  u32x4 g0;
  g0[0] = 1u;                                               // count=1, user mode
  g0[1] = la;                                               // lds_addr (bytes)
  g0[2] = (uint32_t)ga;                                     // global_addr[31:0]
  g0[3] = (uint32_t)((ga >> 32) & 0x1FFFFFFu) | (2u << 30); // addr[56:32] | type=2
  const uint32_t dim0 = (uint32_t)kd;     // tensor_dim0 (elements)
  const uint32_t dim1 = 1u << 20;         // tensor_dim1: large -> no OOB clipping
  const uint32_t st0  = (uint32_t)kd;     // tensor_dim0_stride (elements)
  i32x8 g1;
  g1[0] = (int)((1u << 16)     // data_size = 2 bytes
              | (1u << 20)     // pad_enable
              | (4u << 22)     // pad_interval: 32 DWORDs (=128B row)
              | (3u << 25));   // pad_amount: 4 DWORDs (=16B = LPAD halves)
  g1[1] = (int)((dim0 & 0xFFFFu) << 16);                    // abar=0 | dim0 lo
  g1[2] = (int)(((dim0 >> 16) & 0xFFFFu) | ((dim1 & 0xFFFFu) << 16));
  g1[3] = (int)(((dim1 >> 16) & 0xFFFFu) | ((uint32_t)KTILE << 16)); // tile_dim0
  g1[4] = (int)(uint32_t)rows;                              // tile_dim1 | tile_dim2=0
  g1[5] = (int)st0;                                         // stride0 lo
  g1[6] = 0;                                                // stride0 hi | stride1 lo
  g1[7] = 0;
  i32x4 gz = {0, 0, 0, 0};
#if __clang_major__ >= 23
  i32x8 gz8 = {0, 0, 0, 0, 0, 0, 0, 0};
  __builtin_amdgcn_tensor_load_to_lds(g0, g1, gz, gz, gz8, 0);
#else
  __builtin_amdgcn_tensor_load_to_lds(g0, g1, gz, gz, 0);
#endif
#else
  (void)gsrc; (void)lds_dst; (void)rows; (void)kd;
#endif
}

template <bool HAS_BIAS, bool HAS_RESID, bool DO_GELU, bool DO_MASK,
          bool OUT_F32, bool OUT_BF, bool OUT_BTR>
__global__ __launch_bounds__(128) void gemm_bf16_wmma(
    const bf16_t* __restrict__ A, size_t strideA,   // element strides per batch z
    const bf16_t* __restrict__ W, size_t strideW,
    const float* __restrict__ bias,
    const float* __restrict__ resid, size_t strideR,
    float* __restrict__ outF, size_t strideOF,
    bf16_t* __restrict__ outB, size_t strideOB,
    bf16_t* __restrict__ outT, size_t strideOT, int ldT,
    int N, int Kd, float scale) {
  __shared__ __align__(16) bf16_t As[2][BM][LROW];
  __shared__ __align__(16) bf16_t Bs[2][BN][LROW];

  const int tid = threadIdx.x;
  const int z = blockIdx.z;
  const int tileM = blockIdx.y * BM, tileN = blockIdx.x * BN;

  if (DO_MASK) {
    // causal fast path: block fully above diagonal -> just write -inf
    if (tileN > tileM + (BM - 1)) {
      bf16_t* ob = outB + (size_t)z * strideOB;
      const bf16_t ninf = (bf16_t)(-INFINITY);
      for (int e = tid; e < BM * BN; e += 128) {
        int r = e >> 7, c = e & (BN - 1);
        ob[(size_t)(tileM + r) * N + tileN + c] = ninf;
      }
      return;
    }
  }

  const bf16_t* Ag = A + (size_t)z * strideA + (size_t)tileM * Kd;
  const bf16_t* Wg = W + (size_t)z * strideW + (size_t)tileN * Kd;

  const int lane = tid & 31, wave = tid >> 5;
  const int waveM = (wave & 1) * 32;        // waves {0,1}: rows 0/32
  const int waveN = (wave >> 1) * 64;       // waves {0,1} N-half 0; {2,3} N-half 1
  const int lrow = lane & 15, lhalf = lane >> 4;

  v8f acc[2][4] = {};

  // fill one stage: A via 4 per-lane async b128 chunks, B via one TDM descriptor
  auto issueFill = [&](int st, int kk) {
#pragma unroll
    for (int c = 0; c < 4; ++c) {
      int gc = tid + c * 128;
      int row = gc >> 3, c8 = gc & 7;
      async_b128_to_lds(Ag + (size_t)row * Kd + kk + c8 * 8, &As[st][row][c8 * 8]);
    }
#if HAVE_TDM
    if (wave == 0) tdm_load_tile(Wg + kk, &Bs[st][0][0], BN, Kd);
#else
#pragma unroll
    for (int c = 0; c < 8; ++c) {
      int gc = tid + c * 128;
      int row = gc >> 3, c8 = gc & 7;
      async_b128_to_lds(Wg + (size_t)row * Kd + kk + c8 * 8, &Bs[st][row][c8 * 8]);
    }
#endif
  };
  auto waitFill = [&]() {
    wait_async0();                       // own wave's async A-chunks landed
#if HAVE_TDM
    if (wave == 0) __builtin_amdgcn_s_wait_tensorcnt(0);  // B tile DMA complete
#endif
    __syncthreads();                     // everyone's fills visible
  };

  issueFill(0, 0);
  waitFill();

  const int stages = Kd / KTILE;
  for (int s = 0; s < stages; ++s) {
    const int cur = s & 1;
    if (s + 1 < stages) issueFill(cur ^ 1, (s + 1) * KTILE);
    if (s + 2 < stages) {  // global_prefetch_b8 two stages ahead (GL2 warm-up)
      __builtin_prefetch(Ag + (size_t)(tid >> 3) * Kd + (s + 2) * KTILE + (tid & 7) * 8, 0, 1);
      __builtin_prefetch(Wg + (size_t)(tid >> 3) * Kd + (s + 2) * KTILE + (tid & 7) * 8, 0, 1);
    }
#pragma unroll
    for (int ks = 0; ks < KTILE; ks += 32) {
      // A 16x32 bf16 frag: lanes 0-15 hold K 0..7 & 16..23, lanes 16-31 K 8..15 & 24..31
      v16bf af[2];
#pragma unroll
      for (int i = 0; i < 2; ++i) {
        const int ar = waveM + i * 16 + lrow;
        v8bf lo = *(const v8bf*)&As[cur][ar][ks + 8 * lhalf];
        v8bf hi = *(const v8bf*)&As[cur][ar][ks + 16 + 8 * lhalf];
        af[i] = cat8(lo, hi);
      }
#pragma unroll
      for (int j = 0; j < 4; ++j) {
        // B 32x16 frag: lane n holds column n; lanes 0-15 K 0..7 & 16..23, lanes 16-31 K 8..15 & 24..31
        const int bc = waveN + j * 16 + lrow;
        v8bf lo = *(const v8bf*)&Bs[cur][bc][ks + 8 * lhalf];
        v8bf hi = *(const v8bf*)&Bs[cur][bc][ks + 16 + 8 * lhalf];
        v16bf bfr = cat8(lo, hi);
        acc[0][j] = __builtin_amdgcn_wmma_f32_16x16x32_bf16(
            false, af[0], false, bfr, (short)0, acc[0][j], false, false);
        acc[1][j] = __builtin_amdgcn_wmma_f32_16x16x32_bf16(
            false, af[1], false, bfr, (short)0, acc[1][j], false, false);
      }
    }
    waitFill();
  }

  // epilogue: C layout = 8 VGPRs; lanes 0-15 rows 0-7, lanes 16-31 rows 8-15
#pragma unroll
  for (int i = 0; i < 2; ++i) {
#pragma unroll
    for (int j = 0; j < 4; ++j) {
      const int nCol = tileN + waveN + j * 16 + lrow;
      const float bia = HAS_BIAS ? bias[nCol] : 0.0f;
#pragma unroll
      for (int r = 0; r < 8; ++r) {
        const int row = tileM + waveM + i * 16 + 8 * lhalf + r;
        float v = acc[i][j][r];
        if (DO_MASK) {
          v *= scale;
          if (nCol > row) v = -INFINITY;
        }
        v += bia;
        if (DO_GELU) v = 0.5f * v * (1.0f + erff(v * 0.70710678118654752f));
        if (HAS_RESID) v += resid[(size_t)z * strideR + (size_t)row * N + nCol];
        if (OUT_F32) outF[(size_t)z * strideOF + (size_t)row * N + nCol] = v;
        if (OUT_BF)  outB[(size_t)z * strideOB + (size_t)row * N + nCol] = (bf16_t)v;
        if (OUT_BTR) outT[(size_t)z * strideOT + (size_t)nCol * ldT + row] = (bf16_t)v;
      }
    }
  }
}

// -------------------------------- launch --------------------------------
extern "C" void kernel_launch(void* const* d_in, const int* in_sizes, int n_in,
                              void* d_out, int out_size, void* d_ws, size_t ws_size,
                              hipStream_t stream) {
  const float* x   = (const float*)d_in[0];
  const float* g1  = (const float*)d_in[1];
  const float* be1 = (const float*)d_in[2];
  const float* w1  = (const float*)d_in[3];
  const float* b1  = (const float*)d_in[4];
  const float* wq  = (const float*)d_in[5];
  const float* bq  = (const float*)d_in[6];
  const float* wk  = (const float*)d_in[7];
  const float* bk  = (const float*)d_in[8];
  const float* wv  = (const float*)d_in[9];
  const float* bv  = (const float*)d_in[10];
  const float* w2  = (const float*)d_in[11];
  const float* b2  = (const float*)d_in[12];
  const float* g2  = (const float*)d_in[13];
  const float* be2 = (const float*)d_in[14];
  const float* wf1 = (const float*)d_in[15];
  const float* bf1 = (const float*)d_in[16];
  const float* wf2 = (const float*)d_in[17];
  const float* bf2 = (const float*)d_in[18];

  const size_t DD = (size_t)D_MODEL * D_MODEL;

  char* ws = (char*)d_ws;
  size_t off = 0;
  auto take = [&](size_t bytes) {
    char* p = ws + off;
    off += (bytes + 255) & ~(size_t)255;
    return p;
  };
  bf16_t* w1b  = (bf16_t*)take(DD * 2);
  bf16_t* wqb  = (bf16_t*)take(DD * 2);
  bf16_t* wkb  = (bf16_t*)take(DD * 2);
  bf16_t* wvb  = (bf16_t*)take(DD * 2);
  bf16_t* w2b  = (bf16_t*)take(DD * 2);
  bf16_t* wf1b = (bf16_t*)take(2 * DD * 2);
  bf16_t* wf2b = (bf16_t*)take(2 * DD * 2);
  bf16_t* h1   = (bf16_t*)take((size_t)M8 * D_MODEL * 2);  // LN1 out; reused as LN2 out
  float*  x1f  = (float*) take((size_t)M8 * D_MODEL * 4);
  bf16_t* x1b  = (bf16_t*)take((size_t)M8 * D_MODEL * 2);
  bf16_t* qb   = (bf16_t*)take((size_t)M8 * D_MODEL * 2);
  bf16_t* kb   = (bf16_t*)take((size_t)M8 * D_MODEL * 2);  // contiguous after qb
  bf16_t* vTb  = (bf16_t*)take((size_t)M8 * D_MODEL * 2);  // [B][D][T]
  bf16_t* Sb   = (bf16_t*)take((size_t)BATCH * SEQ_T * SEQ_T * 2);  // scores -> probs
  bf16_t* Ob   = (bf16_t*)take((size_t)M8 * D_MODEL * 2);
  float*  x2f  = (float*) take((size_t)M8 * D_MODEL * 4);
  bf16_t* fb   = qb;  // q/k dead after scores; ff1 activations (8192x2048) overlay q+k

  auto cvt = [&](const float* s, bf16_t* d, size_t n) {
    cvt_f32_bf16<<<dim3((unsigned)((n + 255) / 256)), dim3(256), 0, stream>>>(s, d, (int)n);
  };
  cvt(w1, w1b, DD);
  cvt(wq, wqb, DD);
  cvt(wk, wkb, DD);
  cvt(wv, wvb, DD);
  cvt(w2, w2b, DD);
  cvt(wf1, wf1b, 2 * DD);
  cvt(wf2, wf2b, 2 * DD);

  const dim3 blk(128);
  const size_t TD = (size_t)SEQ_T * D_MODEL;
  const size_t TT = (size_t)SEQ_T * SEQ_T;

  // h1 = LN(x)
  layernorm_bf16<<<dim3(M8), dim3(256), 0, stream>>>(x, g1, be1, h1);

  // x1 = h1 @ w1^T + b1  (f32 + bf16)
  gemm_bf16_wmma<true, false, false, false, true, true, false>
      <<<dim3(D_MODEL / BN, M8 / BM, 1), blk, 0, stream>>>(
          h1, 0, w1b, 0, b1, nullptr, 0, x1f, 0, x1b, 0, nullptr, 0, 0,
          D_MODEL, D_MODEL, 1.f);

  // q = x1 @ wq^T + bq ; k = x1 @ wk^T + bk  (bf16)
  gemm_bf16_wmma<true, false, false, false, false, true, false>
      <<<dim3(D_MODEL / BN, M8 / BM, 1), blk, 0, stream>>>(
          x1b, 0, wqb, 0, bq, nullptr, 0, nullptr, 0, qb, 0, nullptr, 0, 0,
          D_MODEL, D_MODEL, 1.f);
  gemm_bf16_wmma<true, false, false, false, false, true, false>
      <<<dim3(D_MODEL / BN, M8 / BM, 1), blk, 0, stream>>>(
          x1b, 0, wkb, 0, bk, nullptr, 0, nullptr, 0, kb, 0, nullptr, 0, 0,
          D_MODEL, D_MODEL, 1.f);

  // vT[b][d][t] = (x1 @ wv^T + bv)[b][t][d]   (transposed bf16 store, per batch)
  gemm_bf16_wmma<true, false, false, false, false, false, true>
      <<<dim3(D_MODEL / BN, SEQ_T / BM, BATCH), blk, 0, stream>>>(
          x1b, TD, wvb, 0, bv, nullptr, 0, nullptr, 0, nullptr, 0,
          vTb, TD, SEQ_T, D_MODEL, D_MODEL, 1.f);

  // S = causal_mask(q @ k^T / sqrt(D))  (bf16, per batch)
  gemm_bf16_wmma<false, false, false, true, false, true, false>
      <<<dim3(SEQ_T / BN, SEQ_T / BM, BATCH), blk, 0, stream>>>(
          qb, TD, kb, TD, nullptr, nullptr, 0, nullptr, 0, Sb, TT,
          nullptr, 0, 0, SEQ_T, D_MODEL, 0.03125f /* 1/sqrt(1024) */);

  // P = softmax(S) in place
  softmax_rows_bf16<<<dim3(M8), dim3(256), 0, stream>>>(Sb);

  // O = P @ V = P @ (vT)^T   (bf16, per batch)
  gemm_bf16_wmma<false, false, false, false, false, true, false>
      <<<dim3(D_MODEL / BN, SEQ_T / BM, BATCH), blk, 0, stream>>>(
          Sb, TT, vTb, TD, nullptr, nullptr, 0, nullptr, 0, Ob, TD,
          nullptr, 0, 0, D_MODEL, SEQ_T, 1.f);

  // x2 = x1 + O @ w2^T + b2  (f32)
  gemm_bf16_wmma<true, true, false, false, true, false, false>
      <<<dim3(D_MODEL / BN, M8 / BM, 1), blk, 0, stream>>>(
          Ob, 0, w2b, 0, b2, x1f, 0, x2f, 0, nullptr, 0, nullptr, 0, 0,
          D_MODEL, D_MODEL, 1.f);

  // h2 = LN(x2) (reuse h1 buffer)
  layernorm_bf16<<<dim3(M8), dim3(256), 0, stream>>>(x2f, g2, be2, h1);

  // f = gelu(h2 @ wf1^T + bf1)  (bf16, N=2048)
  gemm_bf16_wmma<true, false, true, false, false, true, false>
      <<<dim3(2 * D_MODEL / BN, M8 / BM, 1), blk, 0, stream>>>(
          h1, 0, wf1b, 0, bf1, nullptr, 0, nullptr, 0, fb, 0, nullptr, 0, 0,
          2 * D_MODEL, D_MODEL, 1.f);

  // out = x2 + f @ wf2^T + bf2  (f32 -> d_out)
  gemm_bf16_wmma<true, true, false, false, true, false, false>
      <<<dim3(D_MODEL / BN, M8 / BM, 1), blk, 0, stream>>>(
          fb, 0, wf2b, 0, bf2, x2f, 0, (float*)d_out, 0, nullptr, 0,
          nullptr, 0, 0, D_MODEL, 2 * D_MODEL, 1.f);
}
